// GINSWDSingleModel_85083302133773
// MI455X (gfx1250) — matrix operand order repaired and checked by reading, not compiled, not run on hardware
//
#include <hip/hip_runtime.h>
#include <hip/hip_bf16.h>

// MMD distances: out[b,c] = k_xx[b] + k_yy[c] - 2*k_xy[b,c], gamma = 1/64.
// Inner products via V_WMMA_F32_16X16X32_F16 (f32 accumulate); exp folded to
// a single fma + v_exp_f32 (exp2) per element by pre-scaling norms with
// -gamma*log2(e); LDS rows padded to 144B to kill ds_load_b128 bank conflicts;
// f16 operands staged once in workspace and pulled into LDS with
// global_load_async_to_lds_b128 (ASYNCcnt); tile loops software-pipelined
// 2-deep so next tile's WMMAs fill the WMMA->VALU hazard slots of the
// current tile's exp chain.

#define BS    256
#define NODES 128
#define DIMS  64
#define CATM  64
#define KPTS  32
#define GAMMA (1.0f / 64.0f)
#define LOG2E 1.44269504088896340736f
#define SCALE (-(GAMMA) * LOG2E)     // premultiplied into squared norms
#define LDS_STRIDE 72                // halves; 144B = 9 x 16B chunks, bank-conflict-free

typedef __attribute__((ext_vector_type(16))) _Float16 v16h;
typedef __attribute__((ext_vector_type(8)))  _Float16 v8h;
typedef __attribute__((ext_vector_type(8)))  float    v8f;

#if __has_builtin(__builtin_amdgcn_exp2f)
#define EXP2F(x) __builtin_amdgcn_exp2f(x)
#else
#define EXP2F(x) exp2f(x)
#endif

// ---- WMMA operand loaders from row-major f16 LDS (row stride LDS_STRIDE) ---
// A 16x32 (MxK): lane -> M = lane&15; kh = lane>>4 selects interleaved 8-chunks
// K = {kh*8..kh*8+7} and {16+kh*8..}.
__device__ __forceinline__ v16h load_a_op(const _Float16* base, int row0, int k0) {
  int lane  = threadIdx.x & 31;
  int r     = row0 + (lane & 15);
  int khalf = lane >> 4;
  const _Float16* p = base + r * LDS_STRIDE + k0 + khalf * 8;
  v8h lo = *(const v8h*)p;          // K = k0 + kh*8 + [0..7]
  v8h hi = *(const v8h*)(p + 16);   // K = k0 + 16 + kh*8 + [0..7]
  return __builtin_shufflevector(lo, hi, 0,1,2,3,4,5,6,7,8,9,10,11,12,13,14,15);
}

// B 32x16 (KxN): lane -> N; kh = lane>>4 selects K = {k0+kh*8..+7} and {k0+16+kh*8..+7} (matches A).
// Column n of B = row n of the (row-major) source tile.
__device__ __forceinline__ v16h load_b_op(const _Float16* base, int col0, int k0) {
  int lane  = threadIdx.x & 31;
  int n     = col0 + (lane & 15);
  int khalf = lane >> 4;
  const _Float16* p = base + n * LDS_STRIDE + k0 + khalf * 8;
  v8h lo = *(const v8h*)p;
  v8h hi = *(const v8h*)(p + 16);
  return __builtin_shufflevector(lo, hi, 0,1,2,3,4,5,6,7,8,9,10,11,12,13,14,15);
}

// One 16x16 tile, K=64 as two chained WMMAs against hoisted A operands.
__device__ __forceinline__ v8f tile_mm(const _Float16* bbase, int col0,
                                       v16h a0, v16h a1) {
  v8f c = {};
  v16h b0 = load_b_op(bbase, col0, 0);
  c = __builtin_amdgcn_wmma_f32_16x16x32_f16(false, a0, false, b0,
                                             (short)0, c, false, false);
  v16h b1 = load_b_op(bbase, col0, 32);
  c = __builtin_amdgcn_wmma_f32_16x16x32_f16(false, a1, false, b1,
                                             (short)0, c, false, false);
  return c;
}

// C/D: VGPR v, lanes 0-15: M=v, N=lane; lanes 16-31: M=v+8, N=lane-16.
// rn/cn hold SCALE*||.||^2 so t = rn + cn + C2*dot and exp(-g*d) = exp2(t).
__device__ __forceinline__ float tile_expsum(v8f c, const float* rn, int row0,
                                             const float* cn, int col0) {
  const float C2 = 2.0f * GAMMA * LOG2E;   // == -2*SCALE
  int lane  = threadIdx.x & 31;
  int khalf = lane >> 4;
  float colv = cn[col0 + (lane & 15)];
  float s0 = 0.0f, s1 = 0.0f;              // two chains to shorten dependency
#pragma unroll
  for (int v = 0; v < 8; v += 2) {
    s0 += EXP2F(fmaf(c[v],     C2, rn[row0 + v     + 8 * khalf] + colv));
    s1 += EXP2F(fmaf(c[v + 1], C2, rn[row0 + v + 1 + 8 * khalf] + colv));
  }
  return s0 + s1;
}

__device__ __forceinline__ float wave_reduce(float s) {
#pragma unroll
  for (int m = 16; m >= 1; m >>= 1) s += __shfl_xor(s, m, 32);
  return s;
}

// ---- async global->LDS fill of a padded f16 tile (16B chunks) --------------
__device__ __forceinline__ void async_fill(_Float16* sdst, const _Float16* gsrc,
                                           int nchunks, int nthreads) {
  for (int i = threadIdx.x; i < nchunks; i += nthreads) {
    int r = i >> 3, cc = i & 7;
    unsigned lds = (unsigned)(uintptr_t)(sdst + r * LDS_STRIDE + cc * 8);
    unsigned long long g = (unsigned long long)(uintptr_t)(gsrc + r * DIMS + cc * 8);
    asm volatile("global_load_async_to_lds_b128 %0, %1, off"
                 :: "v"(lds), "v"(g) : "memory");
  }
}

__device__ __forceinline__ void async_wait_barrier() {
  asm volatile("s_wait_asynccnt 0x0" ::: "memory");
  __syncthreads();
}

// ---- prep: f32 -> f16 staging + pre-scaled row squared-norms ---------------
__global__ void prep_kernel(const float* __restrict__ src,
                            _Float16* __restrict__ dsth,
                            float* __restrict__ snrm, int nrows) {
  int i = blockIdx.x * blockDim.x + threadIdx.x;
  if (i >= nrows) return;
  const float4* p = (const float4*)(src + (size_t)i * DIMS);
  v8h* q = (v8h*)(dsth + (size_t)i * DIMS);
  float s = 0.0f;
#pragma unroll
  for (int u = 0; u < 8; ++u) {
    float4 v0 = p[2 * u], v1 = p[2 * u + 1];
    s += v0.x * v0.x + v0.y * v0.y + v0.z * v0.z + v0.w * v0.w;
    s += v1.x * v1.x + v1.y * v1.y + v1.z * v1.z + v1.w * v1.w;
    v8h h = { (_Float16)v0.x, (_Float16)v0.y, (_Float16)v0.z, (_Float16)v0.w,
              (_Float16)v1.x, (_Float16)v1.y, (_Float16)v1.z, (_Float16)v1.w };
    q[u] = h;
  }
  snrm[i] = SCALE * s;
}

// ---- k_yy[c]: one wave per atom, 32x32 Gram = 2x2 tiles x 2 WMMAs ----------
__global__ void kyy_kernel(const _Float16* __restrict__ ah,
                           const float* __restrict__ san,
                           float* __restrict__ kyy) {
  __shared__ __align__(16) _Float16 sA[KPTS * LDS_STRIDE];
  __shared__ float sAn[KPTS];
  int c = blockIdx.x;
  async_fill(sA, ah + (size_t)c * KPTS * DIMS, KPTS * 8, 32);
  sAn[threadIdx.x] = san[c * KPTS + threadIdx.x];
  async_wait_barrier();
  float s = 0.0f;
#pragma unroll
  for (int rt = 0; rt < 2; ++rt) {
    v16h a0 = load_a_op(sA, 16 * rt, 0);
    v16h a1 = load_a_op(sA, 16 * rt, 32);
    v8f c_cur = tile_mm(sA, 0, a0, a1);
#pragma unroll
    for (int ct = 0; ct < 2; ++ct) {
      v8f c_next = {};
      if (ct < 1) c_next = tile_mm(sA, 16 * (ct + 1), a0, a1);
      s += tile_expsum(c_cur, sAn, 16 * rt, sAn, 16 * ct);
      c_cur = c_next;
    }
  }
  s = wave_reduce(s);
  if (threadIdx.x == 0) kyy[c] = s * (1.0f / (KPTS * KPTS));
}

// ---- k_xx[b]: one 8-wave WG per graph, 128x128 Gram ------------------------
__global__ void kxx_kernel(const _Float16* __restrict__ xh,
                           const float* __restrict__ sxn,
                           float* __restrict__ kxx) {
  __shared__ __align__(16) _Float16 sX[NODES * LDS_STRIDE];
  __shared__ float sXn[NODES];
  __shared__ float partial[8];
  int b = blockIdx.x;
  async_fill(sX, xh + (size_t)b * NODES * DIMS, NODES * 8, 256);
  if (threadIdx.x < NODES) sXn[threadIdx.x] = sxn[b * NODES + threadIdx.x];
  async_wait_barrier();
  int w = threadIdx.x >> 5;               // wave id = row block
  v16h a0 = load_a_op(sX, 16 * w, 0);
  v16h a1 = load_a_op(sX, 16 * w, 32);
  float s = 0.0f;
  v8f c_cur = tile_mm(sX, 0, a0, a1);
#pragma unroll
  for (int ct = 0; ct < 8; ++ct) {
    v8f c_next = {};
    if (ct < 7) c_next = tile_mm(sX, 16 * (ct + 1), a0, a1);
    s += tile_expsum(c_cur, sXn, 16 * w, sXn, 16 * ct);
    c_cur = c_next;
  }
  s = wave_reduce(s);
  if ((threadIdx.x & 31) == 0) partial[w] = s;
  __syncthreads();
  if (threadIdx.x == 0) {
    float t = 0.0f;
#pragma unroll
    for (int i = 0; i < 8; ++i) t += partial[i];
    kxx[b] = t * (1.0f / (NODES * NODES));
  }
}

// ---- k_xy + final combine: WG = (graph b, 8-atom group g) ------------------
__global__ void kxy_kernel(const _Float16* __restrict__ xh,
                           const float* __restrict__ sxn,
                           const _Float16* __restrict__ ah,
                           const float* __restrict__ san,
                           const float* __restrict__ kxx,
                           const float* __restrict__ kyy,
                           float* __restrict__ out) {
  __shared__ __align__(16) _Float16 sX[NODES * LDS_STRIDE];      // 18 KB
  __shared__ __align__(16) _Float16 sA[8 * KPTS * LDS_STRIDE];   // 36 KB
  __shared__ float sXn[NODES];
  __shared__ float sAn[8 * KPTS];
  __shared__ float partial[8][8];                                // [wave][atom]
  int b = blockIdx.y;
  int g = blockIdx.x;                                            // atoms 8g..8g+7
  async_fill(sX, xh + (size_t)b * NODES * DIMS, NODES * 8, 256);
  async_fill(sA, ah + (size_t)g * 8 * KPTS * DIMS, 8 * KPTS * 8, 256);
  if (threadIdx.x < NODES) sXn[threadIdx.x] = sxn[b * NODES + threadIdx.x];
  sAn[threadIdx.x] = san[g * 8 * KPTS + threadIdx.x];
  async_wait_barrier();
  int w = threadIdx.x >> 5;               // wave owns row block w (16 rows)
  v16h a0 = load_a_op(sX, 16 * w, 0);
  v16h a1 = load_a_op(sX, 16 * w, 32);
  float acc[8] = {0, 0, 0, 0, 0, 0, 0, 0};
  v8f c_cur = tile_mm(sA, 0, a0, a1);
#pragma unroll
  for (int ct = 0; ct < 16; ++ct) {       // 16 col tiles; atom = ct>>1
    v8f c_next = {};
    if (ct < 15) c_next = tile_mm(sA, 16 * (ct + 1), a0, a1);
    acc[ct >> 1] += tile_expsum(c_cur, sXn, 16 * w, sAn, 16 * ct);
    c_cur = c_next;
  }
#pragma unroll
  for (int a = 0; a < 8; ++a) {
    float s = wave_reduce(acc[a]);
    if ((threadIdx.x & 31) == 0) partial[w][a] = s;
  }
  __syncthreads();
  if (threadIdx.x < 8) {
    int a = threadIdx.x;
    float t = 0.0f;
#pragma unroll
    for (int i = 0; i < 8; ++i) t += partial[i][a];
    float kxy = t * (1.0f / (NODES * KPTS));
    int cg = g * 8 + a;
    out[b * CATM + cg] = kxx[b] + kyy[cg] - 2.0f * kxy;
  }
}

extern "C" void kernel_launch(void* const* d_in, const int* in_sizes, int n_in,
                              void* d_out, int out_size, void* d_ws, size_t ws_size,
                              hipStream_t stream) {
  (void)in_sizes; (void)n_in; (void)out_size; (void)ws_size;
  const float* x     = (const float*)d_in[0];   // [32768, 64] f32
  const float* atoms = (const float*)d_in[1];   // [64, 32, 64] f32 (flat 2048x64)
  float* out = (float*)d_out;                   // [256, 64] f32

  // Workspace (~4.6 MB, L2-resident): xh | ah | sxn | san | kxx | kyy
  char* ws = (char*)d_ws;
  _Float16* xh  = (_Float16*)(ws);              // 32768*64*2 = 4194304 B
  _Float16* ah  = (_Float16*)(ws + 4194304);    //  2048*64*2 =  262144 B
  float*    sxn = (float*)(ws + 4456448);       // 32768*4    =  131072 B
  float*    san = (float*)(ws + 4587520);       //  2048*4    =    8192 B
  float*    kxx = (float*)(ws + 4595712);       //   256*4    =    1024 B
  float*    kyy = (float*)(ws + 4596736);       //    64*4    =     256 B

  prep_kernel<<<dim3((BS * NODES) / 256), dim3(256), 0, stream>>>(x, xh, sxn, BS * NODES);
  prep_kernel<<<dim3((CATM * KPTS) / 256), dim3(256), 0, stream>>>(atoms, ah, san, CATM * KPTS);
  kyy_kernel<<<dim3(CATM), dim3(32), 0, stream>>>(ah, san, kyy);
  kxx_kernel<<<dim3(BS), dim3(256), 0, stream>>>(xh, sxn, kxx);
  kxy_kernel<<<dim3(CATM / 8, BS), dim3(256), 0, stream>>>(xh, sxn, ah, san, kxx, kyy, out);
}
